// ScaledDotProductAttention_11304353923418
// MI455X (gfx1250) — matrix operand order repaired and checked by reading, not compiled, not run on hardware
//
#include <hip/hip_runtime.h>
#include <stdint.h>

typedef __attribute__((ext_vector_type(16))) _Float16 v16h;
typedef __attribute__((ext_vector_type(8)))  float    v8f;
typedef __attribute__((ext_vector_type(4)))  float    v4f;

#define BATCH 32
#define S 2048
#define D 64
#define TEMP_INV 0.125f   // 1 / sqrt(d_k) = 1/8

// Attention for one (batch, 16-query-row) slab per workgroup.
// 256 threads = 8 wave32. Scores slab [16][S] f32 resident in LDS (128KB),
// q tile staged via GLOBAL_LOAD_ASYNC_TO_LDS (ASYNCcnt), attn streamed NT.
__global__ __launch_bounds__(256)
void sdpa_wmma_kernel(const float* __restrict__ q, const float* __restrict__ k,
                      const float* __restrict__ v, const uint8_t* __restrict__ mask,
                      float* __restrict__ out, float* __restrict__ attn) {
  extern __shared__ float sm[];
  float* s_scores = sm;                       // [16][S]    = 131072 B
  float* s_part   = sm + 16 * S;              // [4][32][8] =   4096 B
  float* s_q      = sm + 16 * S + 4 * 32 * 8; // [16][64]   =   4096 B

  const int b    = blockIdx.x >> 7;          // / (S/16)
  const int q0   = (blockIdx.x & 127) << 4;  // query tile base
  const int tid  = threadIdx.x;
  const int wave = tid >> 5;
  const int lane = tid & 31;
  const int lrow = lane & 15;
  // contiguous k split for 16-bit fragments (k = 16h + s), not the ISA table order
  const int a_off = (lane < 16) ? 0 : 16;  // A 16x32: K {0..15} vs {16..31}
  const int b_off = (lane < 16) ? 0 : 16;  // B 32x16: K {0..15} vs {16..31}

  // ---------------- Stage q tile: async global->LDS (one B128 per thread) ----
  {
    const float* qbase = q + ((size_t)b * S + q0) * D;       // 16*64 = 1024 floats
    const uint64_t ga = (uint64_t)(uintptr_t)(qbase + tid * 4);
    const uint32_t la = (uint32_t)(uintptr_t)s_q + tid * 16; // LDS byte address
    asm volatile("global_load_async_to_lds_b128 %0, %1, off"
                 :: "v"(la), "v"(ga) : "memory");
    asm volatile("s_wait_asynccnt 0x0" ::: "memory");
  }
  __syncthreads();

  // ---------------- Phase 1: scores = q @ k^T ----------------
  v16h aq0, aq1;
#pragma unroll
  for (int h = 0; h < 16; ++h) {
    const int c = h;                         // 16 contiguous K -> b128 merges
    aq0[h] = (_Float16)s_q[lrow * D + a_off + c];
    aq1[h] = (_Float16)s_q[lrow * D + a_off + 32 + c];
  }

  for (int i = 0; i < 16; ++i) {             // each wave: 16 of the 128 column tiles
    const int n0 = (wave * 16 + i) << 4;
    const float* krow = k + ((size_t)b * S + n0 + lrow) * D + b_off;
    v16h bk0, bk1;
#pragma unroll
    for (int h = 0; h < 16; ++h) {           // 16 contiguous floats -> b128 merges
      bk0[h] = (_Float16)krow[h];
      bk1[h] = (_Float16)krow[32 + h];
    }
    v8f acc = {};
    acc = __builtin_amdgcn_wmma_f32_16x16x32_f16(false, aq0, false, bk0, (short)0, acc, false, false);
    acc = __builtin_amdgcn_wmma_f32_16x16x32_f16(false, aq1, false, bk1, (short)0, acc, false, false);
    const int mbase = (lane >= 16) ? 8 : 0;  // C/D layout: VGPR r -> M = r (+8 for hi lanes)
#pragma unroll
    for (int r = 0; r < 8; ++r) {
      const int m = mbase + r;
      const int n = n0 + lrow;
      const uint8_t mk =
          __builtin_nontemporal_load(&mask[((size_t)b * S + q0 + m) * S + n]);
      s_scores[m * S + n] = mk ? -__builtin_inff() : acc[r] * TEMP_INV;
    }
  }
  __syncthreads();

  // ---------------- Phase 2: softmax rows + NT-stream attn to HBM ----------------
#pragma unroll
  for (int rr = 0; rr < 2; ++rr) {
    const int row = wave * 2 + rr;
    float* srow = s_scores + row * S;
    float mx = -__builtin_inff();
    for (int j = lane * 4; j < S; j += 128) {          // b128 LDS traffic
      const v4f sv = *(const v4f*)(srow + j);
      mx = fmaxf(mx, fmaxf(fmaxf(sv.x, sv.y), fmaxf(sv.z, sv.w)));
    }
#pragma unroll
    for (int off = 16; off > 0; off >>= 1) mx = fmaxf(mx, __shfl_xor(mx, off, 32));
    float sum = 0.f;
    for (int j = lane * 4; j < S; j += 128) {
      v4f sv = *(v4f*)(srow + j);
      sv.x = __expf(sv.x - mx);
      sv.y = __expf(sv.y - mx);
      sv.z = __expf(sv.z - mx);
      sv.w = __expf(sv.w - mx);
      sum += (sv.x + sv.y) + (sv.z + sv.w);
      *(v4f*)(srow + j) = sv;
    }
#pragma unroll
    for (int off = 16; off > 0; off >>= 1) sum += __shfl_xor(sum, off, 32);
    const float inv = 1.0f / sum;
    float* arow = attn + ((size_t)b * S + q0 + row) * S;
    for (int j = lane * 4; j < S; j += 128) {
      v4f p = *(const v4f*)(srow + j);
      p.x *= inv; p.y *= inv; p.z *= inv; p.w *= inv;
      *(v4f*)(srow + j) = p;                            // keep normalized P for GEMM2
      __builtin_nontemporal_store(p, (v4f*)(arow + j)); // write-once stream: TH=NT
    }
  }
  __syncthreads();

  // ---------------- Phase 3: out = P @ v (K=2048 split across wave pairs) ----------------
  const int tile  = wave & 3;              // 4 output column tiles of 16 (D=64)
  const int khalf = wave >> 2;             // wave pair splits K
  const int d0    = tile << 4;
  v8f acc2 = {};
  for (int kc = 0; kc < 32; ++kc) {
    const int k0 = khalf * 1024 + kc * 32;
    v16h ap, bv;
#pragma unroll
    for (int h = 0; h < 16; ++h) {
      const int c = h;
      ap[h] = (_Float16)s_scores[lrow * S + k0 + a_off + c];
      bv[h] = (_Float16)v[((size_t)b * S + k0 + b_off + h) * D + d0 + lrow];
    }
    acc2 = __builtin_amdgcn_wmma_f32_16x16x32_f16(false, ap, false, bv, (short)0, acc2, false, false);
  }
  if (wave >= 4) {
#pragma unroll
    for (int r = 0; r < 8; ++r) s_part[(tile * 32 + lane) * 8 + r] = acc2[r];
  }
  __syncthreads();
  if (wave < 4) {
    const int mbase = (lane >= 16) ? 8 : 0;
#pragma unroll
    for (int r = 0; r < 8; ++r) {
      const float val = acc2[r] + s_part[(tile * 32 + lane) * 8 + r];
      out[((size_t)b * S + q0 + mbase + r) * D + d0 + lrow] = val;
    }
  }
}

extern "C" void kernel_launch(void* const* d_in, const int* in_sizes, int n_in,
                              void* d_out, int out_size, void* d_ws, size_t ws_size,
                              hipStream_t stream) {
  (void)in_sizes; (void)n_in; (void)out_size; (void)d_ws; (void)ws_size;
  const float*   q    = (const float*)d_in[0];
  const float*   k    = (const float*)d_in[1];
  const float*   v    = (const float*)d_in[2];
  const uint8_t* mask = (const uint8_t*)d_in[3];   // jnp.bool_ = 1 byte/elem

  float* out  = (float*)d_out;                     // [B,S,D] first
  float* attn = out + (size_t)BATCH * S * D;       // then [B,S,S]

  const size_t shmem = (size_t)(16 * S + 4 * 32 * 8 + 16 * D) * sizeof(float); // 139264 B
  dim3 grid(BATCH * (S / 16));   // 4096 workgroups
  dim3 block(256);               // 8 wave32
  sdpa_wmma_kernel<<<grid, block, shmem, stream>>>(q, k, v, mask, out, attn);
}
